// PatchedVisionExpertMLP_29162827940530
// MI455X (gfx1250) — compile-verified, matrix operand order checked by reading, not run on hardware
//
#include <hip/hip_runtime.h>
#include <math.h>

// ---------------------------------------------------------------------------
// PatchedVisionExpertMLP for MI455X (gfx1250), routed-expert bf16 WMMA design
//   B=2, L=2048, D=4096, F=11008, N = B*L = 4096 tokens
// Pipelined double-buffered LDS staging; async global->LDS copies when the
// toolchain exposes the gfx1250 builtins.
// ---------------------------------------------------------------------------

#define PVE_B 2
#define PVE_L 2048
#define PVE_D 4096
#define PVE_F 11008
#define PVE_N (PVE_B * PVE_L)        // 4096 tokens
#define PVE_MAXR 4224                // N + 128 (alignment padding between groups)

#define BM 128
#define BN 128
#define BK 32
#define PITCH 40                     // bf16 elements per LDS row (80B, 16B aligned)
#define TILE_E (BM * PITCH)          // elements per matrix region in LDS

// gfx1250 async global->LDS path, guarded so absence degrades gracefully.
#if defined(__AMDGCN__) && defined(__has_builtin)
#if __has_builtin(__builtin_amdgcn_global_load_async_to_lds_b128) && \
    __has_builtin(__builtin_amdgcn_s_wait_asynccnt)
#define PVE_HAS_ASYNC 1
#endif
#endif
#ifndef PVE_HAS_ASYNC
#define PVE_HAS_ASYNC 0
#endif

#if PVE_HAS_ASYNC
#define PVE_AS_GLOBAL __attribute__((address_space(1)))
#define PVE_AS_LDS    __attribute__((address_space(3)))
#endif

typedef __attribute__((ext_vector_type(16))) __bf16 bf16x16;
typedef __attribute__((ext_vector_type(8)))  __bf16 bf16x8;
typedef __attribute__((ext_vector_type(4)))  __bf16 bf16x4;
typedef __attribute__((ext_vector_type(8)))  float  f32x8;
typedef __attribute__((ext_vector_type(4)))  int    i32x4;

// --- WMMA wrapper: D = A(16x32 bf16) x B(32x16 bf16) + C(16x16 f32) ---------
static __device__ __forceinline__ f32x8 wmma_bf16(bf16x16 a, bf16x16 b, f32x8 c) {
  return __builtin_amdgcn_wmma_f32_16x16x32_bf16(
      /*neg_a=*/false, a, /*neg_b=*/false, b,
      /*c_mod=*/(short)0, c, /*reuse_a=*/false, /*reuse_b=*/false);
}

// Fragment load from an LDS tile stored row-major with pitch PITCH.
// 16-bit A/B layout (ISA 7.12.2): lane<16 holds row m, K {0..7,16..23};
// lane>=16 holds row m, K {8..15,24..31}. Two contiguous 16B LDS loads.
static __device__ __forceinline__ bf16x16
load_frag(const __bf16* tile, int row, int half) {
  const __bf16* q = tile + row * PITCH + half * 8;
  union { bf16x16 v; bf16x8 h[2]; } u;
  u.h[0] = *(const bf16x8*)(q);
  u.h[1] = *(const bf16x8*)(q + 16);
  return u.v;
}

// --- elementwise fp32 -> bf16 convert (hidden states) -----------------------
__global__ __launch_bounds__(256) void pve_convert_f32_bf16(
    const float* __restrict__ in, __bf16* __restrict__ out, int n4) {
  int i = blockIdx.x * 256 + threadIdx.x;
  if (i >= n4) return;
  const float4 v = ((const float4*)in)[i];
  bf16x4 o;
  o.x = (__bf16)v.x; o.y = (__bf16)v.y; o.z = (__bf16)v.z; o.w = (__bf16)v.w;
  ((bf16x4*)out)[i] = o;
}

// --- LDS-tiled transpose + convert: in fp32 [R,C] -> out bf16 [C,R] ---------
__global__ __launch_bounds__(256) void pve_transpose_convert(
    const float* __restrict__ in, __bf16* __restrict__ out, int R, int C) {
  __shared__ float t[32][33];
  const int tx = threadIdx.x & 31, ty = threadIdx.x >> 5;   // 32 x 8
  const int c0 = blockIdx.x * 32, r0 = blockIdx.y * 32;
#pragma unroll
  for (int j = 0; j < 32; j += 8)
    t[ty + j][tx] = in[(size_t)(r0 + ty + j) * C + c0 + tx];
  __syncthreads();
#pragma unroll
  for (int j = 0; j < 32; j += 8)
    out[(size_t)(c0 + ty + j) * R + r0 + tx] = (__bf16)t[tx][ty + j];
}

// --- routing ---------------------------------------------------------------
// meta[0]=cntV  meta[1]=cntL  meta[2]=langBase (cntV rounded to 128)  meta[3]=total
__global__ void pve_zero_meta(int* meta) {
  if (threadIdx.x == 0) { meta[0] = 0; meta[1] = 0; meta[2] = 0; meta[3] = 0; }
}

__global__ __launch_bounds__(256) void pve_route(
    const int* __restrict__ tt, int* meta,
    int* __restrict__ permV, int* __restrict__ permL) {
  int t = blockIdx.x * 256 + threadIdx.x;
  if (t >= PVE_N) return;
  const int l = t & (PVE_L - 1);
  const bool vis = (l < PVE_L - 1) && (tt[t] == 1) && (tt[t + 1] == 1);
  if (vis) {
    int p = atomicAdd(&meta[0], 1);
    permV[p] = t;
  } else {
    int p = atomicAdd(&meta[1], 1);
    permL[p] = t;
  }
}

__global__ void pve_finalize(int* meta) {
  if (threadIdx.x == 0) {
    int lb = ((meta[0] + BM - 1) / BM) * BM;
    meta[2] = lb;
    meta[3] = lb + meta[1];
  }
}

static __device__ __forceinline__ int pve_row_to_token(
    int i, const int* meta, const int* permV, const int* permL) {
  const int cntV = meta[0], cntL = meta[1], lb = meta[2];
  if (i < cntV) return permV[i];
  if (i >= lb && i < lb + cntL) return permL[i - lb];
  return -1;
}

// --- gather bf16 activations by permutation (zero-fill padding rows) --------
__global__ __launch_bounds__(256) void pve_gather(
    const __bf16* __restrict__ xbf, const int* __restrict__ meta,
    const int* __restrict__ permV, const int* __restrict__ permL,
    __bf16* __restrict__ xg) {
  const int i = blockIdx.x;                       // row in [0, MAXR)
  const int tok = pve_row_to_token(i, meta, permV, permL);
  __bf16* dst = xg + (size_t)i * PVE_D;
#pragma unroll
  for (int c = 0; c < 2; ++c) {
    const int e8 = (threadIdx.x + 256 * c) * 8;   // D/8 = 512 chunks
    if (tok >= 0) {
      *(bf16x8*)(dst + e8) = *(const bf16x8*)(xbf + (size_t)tok * PVE_D + e8);
    } else {
      bf16x8 z = {};
      *(bf16x8*)(dst + e8) = z;
    }
  }
}

// --- GEMM1: H = silu(Xg @ Wg^T) * (Xg @ Wu^T), weights pre-transposed [F,D] -
#define G1_CH 6
__global__ __launch_bounds__(256) void pve_gemm_gateup(
    const __bf16* __restrict__ Xg,                 // [MAXR, D]
    const __bf16* __restrict__ WgV, const __bf16* __restrict__ WuV,
    const __bf16* __restrict__ WgL, const __bf16* __restrict__ WuL,
    const int* __restrict__ meta,
    __bf16* __restrict__ H) {                      // [MAXR, F]
  __shared__ __align__(16) __bf16 sBuf[2][3 * TILE_E];

  const int tid = threadIdx.x;
  const int lane = tid & 31, wave = tid >> 5;
  const int m = lane & 15, half = lane >> 4;
  const int rowBase = blockIdx.y * BM;
  const int colBase = blockIdx.x * BN;
  const bool isVision = rowBase < meta[2];
  const __bf16* Wg = isVision ? WgV : WgL;
  const __bf16* Wu = isVision ? WuV : WuL;

  // per-thread staging descriptors: 1536 16B chunks over 256 threads
  const __bf16* srcB[G1_CH];
  int dstOff[G1_CH];
#pragma unroll
  for (int i = 0; i < G1_CH; ++i) {
    const int c = tid + 256 * i;
    const int mat = c >> 9;               // 0:A 1:Wg 2:Wu
    const int loc = c & 511;
    const int row = loc >> 2;
    const int kc = (loc & 3) * 8;
    const __bf16* base = (mat == 0) ? Xg + (size_t)(rowBase + row) * PVE_D
                       : (mat == 1) ? Wg + (size_t)(colBase + row) * PVE_D
                                    : Wu + (size_t)(colBase + row) * PVE_D;
    srcB[i] = base + kc;
    dstOff[i] = mat * TILE_E + row * PITCH + kc;
  }

  f32x8 accG[8], accU[8];
#pragma unroll
  for (int j = 0; j < 8; ++j) { f32x8 z = {}; accG[j] = z; accU[j] = z; }

  auto compute = [&](const __bf16* bufp) {
    const bf16x16 a = load_frag(bufp, wave * 16 + m, half);
#pragma unroll
    for (int j = 0; j < 8; ++j) {
      const bf16x16 bg = load_frag(bufp + TILE_E, j * 16 + m, half);
      accG[j] = wmma_bf16(a, bg, accG[j]);
      const bf16x16 bu = load_frag(bufp + 2 * TILE_E, j * 16 + m, half);
      accU[j] = wmma_bf16(a, bu, accU[j]);
    }
  };

  const int nsteps = PVE_D / BK;   // 128
#if PVE_HAS_ASYNC
  auto aload = [&](int k0, int buf) {
#pragma unroll
    for (int i = 0; i < G1_CH; ++i)
      __builtin_amdgcn_global_load_async_to_lds_b128(
          (PVE_AS_GLOBAL i32x4*)(srcB[i] + k0),
          (PVE_AS_LDS i32x4*)(&sBuf[buf][dstOff[i]]), 0, 0);
  };
  aload(0, 0);
  for (int s = 0; s < nsteps; ++s) {
    __builtin_amdgcn_s_wait_asynccnt(0);   // this wave's current tile landed
    __syncthreads();                        // everyone's tile landed
    if (s + 1 < nsteps) aload((s + 1) * BK, (s + 1) & 1);
    compute(&sBuf[s & 1][0]);               // overlaps next tile's DMA
  }
#else
  bf16x8 regs[G1_CH];
  auto gload = [&](int k0) {
#pragma unroll
    for (int i = 0; i < G1_CH; ++i) {
      regs[i] = *(const bf16x8*)(srcB[i] + k0);
      __builtin_prefetch(srcB[i] + k0 + BK, 0, 1);
    }
  };
  auto lstore = [&](int buf) {
#pragma unroll
    for (int i = 0; i < G1_CH; ++i)
      *(bf16x8*)(&sBuf[buf][dstOff[i]]) = regs[i];
  };
  gload(0);
  lstore(0);
  for (int s = 0; s < nsteps; ++s) {
    if (s + 1 < nsteps) gload((s + 1) * BK);   // loads in flight over compute
    __syncthreads();
    compute(&sBuf[s & 1][0]);
    if (s + 1 < nsteps) lstore((s + 1) & 1);
  }
#endif

  // epilogue: silu(g)*u -> bf16 H.  C layout: n = j*16 + m; mrow = r + half*8
#pragma unroll
  for (int j = 0; j < 8; ++j) {
#pragma unroll
    for (int r = 0; r < 8; ++r) {
      const float g = accG[j][r];
      const float u = accU[j][r];
      const float h = (g / (1.0f + __expf(-g))) * u;
      const int mrow = rowBase + wave * 16 + r + half * 8;
      const int n = colBase + j * 16 + m;
      H[(size_t)mrow * PVE_F + n] = (__bf16)h;
    }
  }
}

// --- GEMM2: out[token] = H @ Wd^T, scatter fp32 rows by permutation ---------
#define G2_CH 4
__global__ __launch_bounds__(256) void pve_gemm_down(
    const __bf16* __restrict__ H,                  // [MAXR, F]
    const __bf16* __restrict__ WdV,                // [D, F] (transposed)
    const __bf16* __restrict__ WdL,
    const int* __restrict__ meta,
    const int* __restrict__ permV, const int* __restrict__ permL,
    float* __restrict__ out) {                     // [N, D]
  __shared__ __align__(16) __bf16 sBuf[2][2 * TILE_E];

  const int tid = threadIdx.x;
  const int lane = tid & 31, wave = tid >> 5;
  const int m = lane & 15, half = lane >> 4;
  const int rowBase = blockIdx.y * BM;
  const int colBase = blockIdx.x * BN;
  const bool isVision = rowBase < meta[2];
  const __bf16* Wd = isVision ? WdV : WdL;

  const __bf16* srcB[G2_CH];
  int dstOff[G2_CH];
#pragma unroll
  for (int i = 0; i < G2_CH; ++i) {
    const int c = tid + 256 * i;
    const int mat = c >> 9;
    const int loc = c & 511;
    const int row = loc >> 2;
    const int kc = (loc & 3) * 8;
    const __bf16* base = (mat == 0) ? H + (size_t)(rowBase + row) * PVE_F
                                    : Wd + (size_t)(colBase + row) * PVE_F;
    srcB[i] = base + kc;
    dstOff[i] = mat * TILE_E + row * PITCH + kc;
  }

  f32x8 acc[8];
#pragma unroll
  for (int j = 0; j < 8; ++j) { f32x8 z = {}; acc[j] = z; }

  auto compute = [&](const __bf16* bufp) {
    const bf16x16 a = load_frag(bufp, wave * 16 + m, half);
#pragma unroll
    for (int j = 0; j < 8; ++j) {
      const bf16x16 b = load_frag(bufp + TILE_E, j * 16 + m, half);
      acc[j] = wmma_bf16(a, b, acc[j]);
    }
  };

  const int nsteps = PVE_F / BK;   // 344
#if PVE_HAS_ASYNC
  auto aload = [&](int k0, int buf) {
#pragma unroll
    for (int i = 0; i < G2_CH; ++i)
      __builtin_amdgcn_global_load_async_to_lds_b128(
          (PVE_AS_GLOBAL i32x4*)(srcB[i] + k0),
          (PVE_AS_LDS i32x4*)(&sBuf[buf][dstOff[i]]), 0, 0);
  };
  aload(0, 0);
  for (int s = 0; s < nsteps; ++s) {
    __builtin_amdgcn_s_wait_asynccnt(0);
    __syncthreads();
    if (s + 1 < nsteps) aload((s + 1) * BK, (s + 1) & 1);
    compute(&sBuf[s & 1][0]);
  }
#else
  bf16x8 regs[G2_CH];
  auto gload = [&](int k0) {
#pragma unroll
    for (int i = 0; i < G2_CH; ++i) {
      regs[i] = *(const bf16x8*)(srcB[i] + k0);
      __builtin_prefetch(srcB[i] + k0 + BK, 0, 1);
    }
  };
  auto lstore = [&](int buf) {
#pragma unroll
    for (int i = 0; i < G2_CH; ++i)
      *(bf16x8*)(&sBuf[buf][dstOff[i]]) = regs[i];
  };
  gload(0);
  lstore(0);
  for (int s = 0; s < nsteps; ++s) {
    if (s + 1 < nsteps) gload((s + 1) * BK);
    __syncthreads();
    compute(&sBuf[s & 1][0]);
    if (s + 1 < nsteps) lstore((s + 1) & 1);
  }
#endif

  // scatter epilogue: row -> original token; padding rows dropped
  int tok[8];
#pragma unroll
  for (int r = 0; r < 8; ++r)
    tok[r] = pve_row_to_token(rowBase + wave * 16 + r + half * 8,
                              meta, permV, permL);
#pragma unroll
  for (int j = 0; j < 8; ++j) {
#pragma unroll
    for (int r = 0; r < 8; ++r) {
      if (tok[r] >= 0)
        out[(size_t)tok[r] * PVE_D + colBase + j * 16 + m] = acc[j][r];
    }
  }
}

// ---------------------------------------------------------------------------
extern "C" void kernel_launch(void* const* d_in, const int* in_sizes, int n_in,
                              void* d_out, int out_size, void* d_ws, size_t ws_size,
                              hipStream_t stream) {
  (void)in_sizes; (void)n_in; (void)out_size; (void)ws_size;

  const float* x      = (const float*)d_in[0];   // [B,L,D] fp32
  const int*   tt     = (const int*)d_in[1];     // [B,L]
  const float* w_gv   = (const float*)d_in[2];   // [D,F]
  const float* w_uv   = (const float*)d_in[3];   // [D,F]
  const float* w_dv   = (const float*)d_in[4];   // [F,D]
  const float* w_gl   = (const float*)d_in[5];   // [D,F]
  const float* w_ul   = (const float*)d_in[6];   // [D,F]
  const float* w_dl   = (const float*)d_in[7];   // [F,D]
  float* out = (float*)d_out;

  // workspace carve-up (all sizes multiples of 256B)
  char* ws = (char*)d_ws;
  const size_t szX  = (size_t)PVE_N * PVE_D * 2;          // 32 MB  bf16 x
  const size_t szW  = (size_t)PVE_D * PVE_F * 2;          // 90 MB  per weight
  const size_t szXG = (size_t)PVE_MAXR * PVE_D * 2;       // gathered x
  const size_t szH  = (size_t)PVE_MAXR * PVE_F * 2;       // hidden
  const size_t szP  = (size_t)PVE_N * sizeof(int);

  __bf16* xbf  = (__bf16*)(ws);                 ws += szX;
  __bf16* WgV  = (__bf16*)(ws);                 ws += szW;   // [F,D]
  __bf16* WuV  = (__bf16*)(ws);                 ws += szW;   // [F,D]
  __bf16* WdV  = (__bf16*)(ws);                 ws += szW;   // [D,F]
  __bf16* WgL  = (__bf16*)(ws);                 ws += szW;
  __bf16* WuL  = (__bf16*)(ws);                 ws += szW;
  __bf16* WdL  = (__bf16*)(ws);                 ws += szW;
  __bf16* xg   = (__bf16*)(ws);                 ws += szXG;
  __bf16* H    = (__bf16*)(ws);                 ws += szH;
  int* permV   = (int*)(ws);                    ws += szP;
  int* permL   = (int*)(ws);                    ws += szP;
  int* meta    = (int*)(ws);

  // 1) reset routing state (must be re-zeroed every call)
  pve_zero_meta<<<1, 32, 0, stream>>>(meta);

  // 2) activations fp32 -> bf16
  {
    const int n4 = PVE_N * PVE_D / 4;
    pve_convert_f32_bf16<<<(n4 + 255) / 256, 256, 0, stream>>>(x, xbf, n4);
  }

  // 3) weights fp32 -> bf16 with transpose to [N_out, K] layout
  {
    dim3 gDF(PVE_F / 32, PVE_D / 32);   // in [D,F] -> out [F,D]
    dim3 gFD(PVE_D / 32, PVE_F / 32);   // in [F,D] -> out [D,F]
    pve_transpose_convert<<<gDF, 256, 0, stream>>>(w_gv, WgV, PVE_D, PVE_F);
    pve_transpose_convert<<<gDF, 256, 0, stream>>>(w_uv, WuV, PVE_D, PVE_F);
    pve_transpose_convert<<<gFD, 256, 0, stream>>>(w_dv, WdV, PVE_F, PVE_D);
    pve_transpose_convert<<<gDF, 256, 0, stream>>>(w_gl, WgL, PVE_D, PVE_F);
    pve_transpose_convert<<<gDF, 256, 0, stream>>>(w_ul, WuL, PVE_D, PVE_F);
    pve_transpose_convert<<<gFD, 256, 0, stream>>>(w_dl, WdL, PVE_F, PVE_D);
  }

  // 4) route tokens by vision mask, 5) compute aligned language base
  pve_route<<<PVE_N / 256, 256, 0, stream>>>(tt, meta, permV, permL);
  pve_finalize<<<1, 32, 0, stream>>>(meta);

  // 6) gather activations into expert-grouped rows (zero padding rows)
  pve_gather<<<PVE_MAXR, 256, 0, stream>>>(xbf, meta, permV, permL, xg);

  // 7) fused gate/up SwiGLU GEMM -> H
  {
    dim3 grid(PVE_F / BN, PVE_MAXR / BM);   // 86 x 33
    pve_gemm_gateup<<<grid, 256, 0, stream>>>(xg, WgV, WuV, WgL, WuL, meta, H);
  }

  // 8) down-projection GEMM with scatter to d_out
  {
    dim3 grid(PVE_D / BN, PVE_MAXR / BM);   // 32 x 33
    pve_gemm_down<<<grid, 256, 0, stream>>>(H, WdV, WdL, meta, permV, permL, out);
  }
}
